// ConvGRUBandCell2d_85950885528364
// MI455X (gfx1250) — matrix operand order repaired and checked by reading, not compiled, not run on hardware
//
#include <hip/hip_runtime.h>
#include <hip/hip_bf16.h>

#define BATCH 64
#define CCH   512
#define FREQ  1024
#define NPIX  (BATCH * FREQ)      // 65536 pixels
#define EPS   1e-6f

typedef __attribute__((ext_vector_type(16))) __bf16 v16bf;
typedef __attribute__((ext_vector_type(8)))  __bf16 v8bf;
typedef __attribute__((ext_vector_type(8)))  float  v8f;

__device__ __forceinline__ float sigmoidf_(float x) {
    return 1.0f / (1.0f + __expf(-x));
}

// ---------------------------------------------------------------------------
// Kernel 0: fp32 -> bf16 weight conversion (done once, not per GEMM block).
// ---------------------------------------------------------------------------
__global__ __launch_bounds__(256)
void k_cvt_w(const float* __restrict__ src, __bf16* __restrict__ dst) {
    const size_t i = (size_t)(blockIdx.x * blockDim.x + threadIdx.x) * 8;
    const float* s = src + i;
    v8bf d;
#pragma unroll
    for (int j = 0; j < 8; ++j) d[j] = (__bf16)s[j];
    *(v8bf*)(dst + i) = d;
}

// ---------------------------------------------------------------------------
// Kernel 1: per-pixel RMSNorm inverse factors for x_t and h_prev.
// ---------------------------------------------------------------------------
__global__ __launch_bounds__(256)
void k_rms_inv(const float* __restrict__ x, const float* __restrict__ h,
               float* __restrict__ inv_x, float* __restrict__ inv_h) {
    const int p = blockIdx.x * blockDim.x + threadIdx.x;
    const int b = p >> 10;            // / FREQ
    const int f = p & (FREQ - 1);
    const float* xp = x + ((size_t)b * CCH) * FREQ + f;
    const float* hp = h + ((size_t)b * CCH) * FREQ + f;
    float sx = 0.0f, sh = 0.0f;
#pragma unroll 4
    for (int c = 0; c < CCH; ++c) {
        float vx = xp[(size_t)c * FREQ];
        float vh = hp[(size_t)c * FREQ];
        sx += vx * vx;
        sh += vh * vh;
    }
    inv_x[p] = rsqrtf(sx * (1.0f / CCH) + EPS);
    inv_h[p] = rsqrtf(sh * (1.0f / CCH) + EPS);
}

// ---------------------------------------------------------------------------
// Kernel 2: build bf16 A-matrices (pixel-major [p, c]).
//   Ax[p,c] = x * inv_x * w_in[c]
//   Ah[p,c] = depthwise K=3 band conv (same pad, along f) of rmsnormed h + bias
// ---------------------------------------------------------------------------
__global__ __launch_bounds__(256)
void k_build_A(const float* __restrict__ x, const float* __restrict__ h,
               const float* __restrict__ w_in, const float* __restrict__ w_hid,
               const float* __restrict__ hmixW, const float* __restrict__ hmixb,
               const float* __restrict__ inv_x, const float* __restrict__ inv_h,
               __bf16* __restrict__ Ax, __bf16* __restrict__ Ah) {
    const int p = blockIdx.x * blockDim.x + threadIdx.x;
    const int b = p >> 10;
    const int f = p & (FREQ - 1);
    const float ivx  = inv_x[p];
    const float ivh  = inv_h[p];
    const float ivhm = (f > 0)        ? inv_h[p - 1] : 0.0f;
    const float ivhp = (f < FREQ - 1) ? inv_h[p + 1] : 0.0f;
    const float* xp = x + ((size_t)b * CCH) * FREQ + f;
    const float* hp = h + ((size_t)b * CCH) * FREQ + f;
    __bf16* axp = Ax + (size_t)p * CCH;
    __bf16* ahp = Ah + (size_t)p * CCH;

    for (int c0 = 0; c0 < CCH; c0 += 8) {
        v8bf ax, ah;
#pragma unroll
        for (int j = 0; j < 8; ++j) {
            const int c = c0 + j;
            const size_t off = (size_t)c * FREQ;
            ax[j] = (__bf16)(xp[off] * ivx * w_in[c]);
            float hm = (f > 0)        ? hp[off - 1] * ivhm : 0.0f;
            float hc =                  hp[off]     * ivh;
            float hq = (f < FREQ - 1) ? hp[off + 1] * ivhp : 0.0f;
            float y = w_hid[c] * (hmixW[c * 3 + 0] * hm +
                                  hmixW[c * 3 + 1] * hc +
                                  hmixW[c * 3 + 2] * hq) + hmixb[c];
            ah[j] = (__bf16)y;
        }
        *(v8bf*)(axp + c0) = ax;
        *(v8bf*)(ahp + c0) = ah;
    }
}

// ---------------------------------------------------------------------------
// WMMA helpers (CDNA5 wave32, v_wmma_f32_16x16x32_bf16)
// 16-bit A-matrix 16x32: lane L holds row M=L%16; values are
// K = [16*half .. +15] in this code's contiguous order, half = L/16.
// B-matrix 32x16: lane L holds column N=L%16; K = [16*half .. +15], same order as A.
// ---------------------------------------------------------------------------
__device__ __forceinline__ v16bf load_a_bf16(const __bf16* __restrict__ base,
                                             int row, int k0, int half) {
    const __bf16* pr = base + (size_t)row * CCH + k0 + 16 * half;
    v8bf lo = *(const v8bf*)pr;
    v8bf hi = *(const v8bf*)(pr + 8);
    v16bf a;
#pragma unroll
    for (int i = 0; i < 8; ++i) { a[i] = lo[i]; a[8 + i] = hi[i]; }
    return a;
}

__device__ __forceinline__ v8f wmma_bf16(v16bf a, v16bf b, v8f c) {
    return __builtin_amdgcn_wmma_f32_16x16x32_bf16(
        /*neg_a=*/false, a, /*neg_b=*/false, b,
        /*c_mod=*/(short)0, c, /*reuse_a=*/false, /*reuse_b=*/false);
}

// Async global->LDS copy of 8 bytes (GLOBAL_LOAD_ASYNC_TO_LDS_B64, ASYNCcnt).
// lds_off = LDS byte address (low 32 bits of the generic shared pointer);
// gptr    = 64-bit global address.
__device__ __forceinline__ void async_g2l_b64(uint32_t lds_off, const void* gptr) {
    asm volatile("global_load_async_to_lds_b64 %0, %1, off"
                 :: "v"(lds_off), "v"(gptr) : "memory");
}
__device__ __forceinline__ void wait_asynccnt0() {
    asm volatile("s_wait_asynccnt 0x0" ::: "memory");
}

// ---------------------------------------------------------------------------
// Kernel 3: fused dual-GEMM + GRU gates.
// Block = 256 threads = 8 wave32s. Tile: 64 pixels x 32 gate channels.
// B tiles (6 of them: xW/hW x {r,z,n} chunks) staged in LDS per K-step via
// async global->LDS copies (ASYNCcnt), shared by all waves.
// ---------------------------------------------------------------------------
#define BROW 40   // padded LDS row length (32 data + 8 pad bf16), 80B rows

__global__ __launch_bounds__(256)
void k_gemm_gates(const __bf16* __restrict__ Ax, const __bf16* __restrict__ Ah,
                  const __bf16* __restrict__ xWb, const __bf16* __restrict__ hWb,
                  const float* __restrict__ xb,  const float* __restrict__ hb,
                  const float* __restrict__ x_t, const float* __restrict__ h_prev,
                  float* __restrict__ outT) {
    __shared__ __bf16 lb[6][32][BROW];   // 6 B-tiles, [n][k], padded rows

    const int lane = threadIdx.x & 31;
    const int wave = threadIdx.x >> 5;
    const int wi   = wave >> 1;                  // pixel sub-tile 0..3
    const int wj   = wave & 1;                   // channel sub-tile 0..1
    const int p0   = blockIdx.x * 64 + wi * 16;  // pixel tile base (one batch)
    const int nb0  = blockIdx.y * 32;            // block channel base in [0,512)
    const int nb   = nb0 + wj * 16;
    const int half = lane >> 4;
    const int mrow = lane & 15;

    // Cooperative-stage indexing: thread covers 4 bf16 (8B) of one tile row.
    const int sn = threadIdx.x >> 3;             // row n: 0..31
    const int sj = (threadIdx.x & 7) * 4;        // k offset within row
    const __bf16* srcs[6] = {
        xWb + (size_t)(nb0)        * CCH, xWb + (size_t)(nb0 + 512)  * CCH,
        xWb + (size_t)(nb0 + 1024) * CCH, hWb + (size_t)(nb0)        * CCH,
        hWb + (size_t)(nb0 + 512)  * CCH, hWb + (size_t)(nb0 + 1024) * CCH };

    v8f aXR = {}, aXZ = {}, aXN = {}, aHR = {}, aHZ = {}, aHN = {};

    const __bf16* axRow = Ax + (size_t)(p0 + mrow) * CCH;
    const __bf16* ahRow = Ah + (size_t)(p0 + mrow) * CCH;
    const int bn = wj * 16 + mrow;               // lane's column within block tile

    for (int k0 = 0; k0 < CCH; k0 += 32) {
        // ---- stage 6 B-tiles into LDS via async copies (6 x 8B / thread) ----
#pragma unroll
        for (int t = 0; t < 6; ++t) {
            const __bf16* g = srcs[t] + (size_t)sn * CCH + k0 + sj;
            uint32_t loff = (uint32_t)(uintptr_t)&lb[t][sn][sj];
            async_g2l_b64(loff, (const void*)g);
        }
        wait_asynccnt0();
        __syncthreads();

        // ---- prefetch next K-step A rows ----
        if (k0 + 32 < CCH) {
            __builtin_prefetch(axRow + k0 + 32, 0, 1);
            __builtin_prefetch(ahRow + k0 + 32, 0, 1);
        }

        // ---- fragments + WMMA ----
        v16bf ax = load_a_bf16(Ax, p0 + mrow, k0, half);
        v16bf ah = load_a_bf16(Ah, p0 + mrow, k0, half);
        v16bf bXR = *(const v16bf*)&lb[0][bn][16 * half];
        v16bf bXZ = *(const v16bf*)&lb[1][bn][16 * half];
        v16bf bXN = *(const v16bf*)&lb[2][bn][16 * half];
        v16bf bHR = *(const v16bf*)&lb[3][bn][16 * half];
        v16bf bHZ = *(const v16bf*)&lb[4][bn][16 * half];
        v16bf bHN = *(const v16bf*)&lb[5][bn][16 * half];

        aXR = wmma_bf16(ax, bXR, aXR);
        aXZ = wmma_bf16(ax, bXZ, aXZ);
        aXN = wmma_bf16(ax, bXN, aXN);
        aHR = wmma_bf16(ah, bHR, aHR);
        aHZ = wmma_bf16(ah, bHZ, aHZ);
        aHN = wmma_bf16(ah, bHN, aHN);
        __syncthreads();
    }

    // Epilogue: lane owns column N = lane%16 (gate channel), rows M = r+8*half.
    const int   cch   = nb + mrow;
    const int   b     = p0 >> 10;
    const int   fbase = p0 & (FREQ - 1);
    const float xbr = xb[cch], xbz = xb[cch + 512], xbn = xb[cch + 1024];
    const float hbr = hb[cch], hbz = hb[cch + 512], hbn = hb[cch + 1024];
    const size_t rowoff = ((size_t)b * CCH + cch) * FREQ + fbase;

#pragma unroll
    for (int r = 0; r < 8; ++r) {
        const int m = r + 8 * half;
        const size_t idx = rowoff + m;
        float rg = sigmoidf_(aXR[r] + xbr + aHR[r] + hbr);
        float zg = sigmoidf_(aXZ[r] + xbz + aHZ[r] + hbz);
        float cd = tanhf(aXN[r] + xbn + rg * (aHN[r] + hbn));
        float hp = h_prev[idx];
        float hn = (1.0f - zg) * cd + zg * hp;
        outT[idx] = hn + x_t[idx];           // pre-norm result
    }
}

// ---------------------------------------------------------------------------
// Kernel 4: in-place output RMSNorm over channels.
// ---------------------------------------------------------------------------
__global__ __launch_bounds__(256)
void k_out_rmsnorm(float* __restrict__ out, const float* __restrict__ w_out) {
    const int p = blockIdx.x * blockDim.x + threadIdx.x;
    const int b = p >> 10;
    const int f = p & (FREQ - 1);
    float* op = out + ((size_t)b * CCH) * FREQ + f;
    float s = 0.0f;
#pragma unroll 4
    for (int c = 0; c < CCH; ++c) {
        float v = op[(size_t)c * FREQ];
        s += v * v;
    }
    const float inv = rsqrtf(s * (1.0f / CCH) + EPS);
    for (int c = 0; c < CCH; ++c) {
        op[(size_t)c * FREQ] *= inv * w_out[c];
    }
}

// ---------------------------------------------------------------------------
// Launch
// ---------------------------------------------------------------------------
extern "C" void kernel_launch(void* const* d_in, const int* in_sizes, int n_in,
                              void* d_out, int out_size, void* d_ws, size_t ws_size,
                              hipStream_t stream) {
    const float* x_t    = (const float*)d_in[0];
    const float* h_prev = (const float*)d_in[1];
    const float* w_in   = (const float*)d_in[2];
    const float* w_hid  = (const float*)d_in[3];
    const float* w_out  = (const float*)d_in[4];
    const float* xW     = (const float*)d_in[5];
    const float* xb     = (const float*)d_in[6];
    const float* hmixW  = (const float*)d_in[7];
    const float* hmixb  = (const float*)d_in[8];
    const float* hW     = (const float*)d_in[9];
    const float* hb     = (const float*)d_in[10];
    float* out = (float*)d_out;

    // Workspace: inv_x | inv_h | Ax(bf16) | Ah(bf16) | xWb(bf16) | hWb(bf16)
    float*  inv_x = (float*)d_ws;
    float*  inv_h = inv_x + NPIX;
    __bf16* Ax    = (__bf16*)(inv_h + NPIX);
    __bf16* Ah    = Ax + (size_t)NPIX * CCH;
    __bf16* xWb   = Ah + (size_t)NPIX * CCH;
    __bf16* hWb   = xWb + (size_t)3 * CCH * CCH;

    const int wElems = 3 * CCH * CCH;            // 786432 per weight matrix

    // 0) Weight conversion (once)
    k_cvt_w<<<wElems / 8 / 256, 256, 0, stream>>>(xW, xWb);
    k_cvt_w<<<wElems / 8 / 256, 256, 0, stream>>>(hW, hWb);

    // 1) RMSNorm inverse factors
    k_rms_inv<<<NPIX / 256, 256, 0, stream>>>(x_t, h_prev, inv_x, inv_h);

    // 2) Build bf16 activation matrices (norm + band conv fused)
    k_build_A<<<NPIX / 256, 256, 0, stream>>>(x_t, h_prev, w_in, w_hid,
                                              hmixW, hmixb, inv_x, inv_h, Ax, Ah);

    // 3) Fused dual-GEMM + gates
    dim3 ggrid(NPIX / 64, CCH / 32);
    k_gemm_gates<<<ggrid, 256, 0, stream>>>(Ax, Ah, xWb, hWb, xb, hb,
                                            x_t, h_prev, out);

    // 4) Output RMSNorm in place
    k_out_rmsnorm<<<NPIX / 256, 256, 0, stream>>>(out, w_out);
}
